// SchnetInteraction_23218593202759
// MI455X (gfx1250) — matrix operand order repaired and checked by reading, not compiled, not run on hardware
//
#include <hip/hip_runtime.h>
#include <hip/hip_bf16.h>

// MI455X / gfx1250, wave32. WMMA f32_16x16x32_f16 for all GEMMs,
// TDM (tensor_load_to_lds) for weight staging in the hot edge kernel.
// SchNet interaction block: B=4, N=256, DN=128, DE=64, DF=128.

typedef _Float16 v16h __attribute__((ext_vector_type(16)));
typedef _Float16 v8h  __attribute__((ext_vector_type(8)));
typedef float    v8f  __attribute__((ext_vector_type(8)));
typedef unsigned int u32x4 __attribute__((ext_vector_type(4)));
typedef int          i32x4 __attribute__((ext_vector_type(4)));
typedef int          i32x8 __attribute__((ext_vector_type(8)));

#define WMMA_F16(A, B, C) \
    __builtin_amdgcn_wmma_f32_16x16x32_f16(false, (A), false, (B), (short)0, (C), false, false)

static constexpr int Bn = 4, Nn = 256, DN = 128, DE = 64, DF = 128;
static constexpr int ROWS = Bn * Nn; // 1024

#if defined(__has_builtin)
#  if __has_builtin(__builtin_amdgcn_tensor_load_to_lds) && \
      __has_builtin(__builtin_amdgcn_s_wait_tensorcnt)
#    define HAVE_TDM 1
#  endif
#endif
#ifndef HAVE_TDM
#  define HAVE_TDM 0
#endif

// ---------------- fragment builders (ISA 7.12.2 layouts) ----------------
__device__ inline v16h frag_a_h(const _Float16* base, int rs, int lane, int kbase) {
    const _Float16* p = base + (lane & 15) * rs + kbase + ((lane >> 4) << 3);
    v8h lo = *(const v8h*)(p);
    v8h hi = *(const v8h*)(p + 16);
    v16h r;
#pragma unroll
    for (int i = 0; i < 8; ++i) { r[i] = lo[i]; r[i + 8] = hi[i]; }
    return r;
}

__device__ inline v16h frag_a_f32(const float* base, int rs, int lane, int kbase) {
    const float* p = base + (lane & 15) * rs + kbase + ((lane >> 4) << 3);
    v16h r;
#pragma unroll
    for (int i = 0; i < 8; ++i) {
        r[i]     = (_Float16)p[i];
        r[i + 8] = (_Float16)p[16 + i];
    }
    return r;
}

__device__ inline v16h frag_b_h(const _Float16* wT, int kd, int lane, int n0, int kbase) {
    const _Float16* p = wT + (n0 + (lane & 15)) * kd + kbase + ((lane >> 4) << 3);
    v8h lo = *(const v8h*)(p);
    v8h hi = *(const v8h*)(p + 16);
    v16h r;
#pragma unroll
    for (int i = 0; i < 8; ++i) { r[i] = lo[i]; r[i + 8] = hi[i]; }
    return r;
}

__device__ inline float silu_f(float x) { return x / (1.0f + __expf(-x)); }

#if HAVE_TDM
// LDS byte offset of a shared-memory pointer (strip the flat aperture).
__device__ inline unsigned lds_offset(const void* p) {
    return (unsigned)(unsigned long long)(const __attribute__((address_space(3))) void*)p;
}
// Issue a 1-row TDM copy of n8 8-byte units global -> LDS (D# per ISA ch.8).
__device__ inline void tdm_load_1d(unsigned lds_off, const void* gsrc, unsigned n8) {
    unsigned long long ga = (unsigned long long)gsrc;
    u32x4 g0;
    g0[0] = 1u;                                      // count=1, user mode
    g0[1] = lds_off;                                 // lds_addr @ [63:32]
    g0[2] = (unsigned)ga;                            // global_addr[31:0]
    g0[3] = (unsigned)((ga >> 32) & 0x01FFFFFFu) | 0x80000000u; // addr[56:32] | type=2
    i32x8 g1;
    g1[0] = 0x00030000;                              // mask=0, data_size=3 (8B units)
    g1[1] = (int)((n8 & 0xFFFFu) << 16);             // tensor_dim0[15:0] @ [63:48]
    g1[2] = (int)(((n8 >> 16) & 0xFFFFu) | (1u << 16)); // dim0[31:16] | tensor_dim1=1
    g1[3] = (int)((n8 & 0xFFFFu) << 16);             // tile_dim0 @ [127:112]
    g1[4] = 1;                                       // tile_dim1=1, tile_dim2=0
    g1[5] = (int)n8;                                 // tensor_dim0_stride[31:0]
    g1[6] = 0;
    g1[7] = 0;
    i32x4 z4 = {0, 0, 0, 0};
#if defined(__clang_major__) && __clang_major__ >= 23
    i32x8 z8 = {0, 0, 0, 0, 0, 0, 0, 0};
    __builtin_amdgcn_tensor_load_to_lds(g0, g1, z4, z4, z8, 0);
#else
    __builtin_amdgcn_tensor_load_to_lds(g0, g1, z4, z4, 0);
#endif
}
#endif

// ---------------- kernel 0: one-time weight transpose + fp32->f16 ----------------
// dst[n*K + k] = (f16) src[k*N + n]
__global__ __launch_bounds__(256) void k_cvt_t(const float* __restrict__ src,
                                               _Float16* __restrict__ dst,
                                               int K, int N) {
    int idx = blockIdx.x * 256 + threadIdx.x;
    if (idx < K * N) {
        int n = idx / K, k = idx - n * K;
        dst[idx] = (_Float16)src[k * N + n];
    }
}

// ---------------- kernel 1: X = node_vec @ bcW + bcb ----------------
__global__ __launch_bounds__(256) void k_bc(const float* __restrict__ node_vec,
                                            const _Float16* __restrict__ hBcT,
                                            const float* __restrict__ bcb,
                                            float* __restrict__ X) {
    __shared__ alignas(16) _Float16 sWT[DF * DN];
    __shared__ alignas(16) _Float16 sA[16 * DN];
    const int tid = threadIdx.x, lane = tid & 31, wv = tid >> 5;
    const int m0 = blockIdx.x * 16;

    for (int idx = tid; idx < (DF * DN) / 8; idx += 256)
        ((uint4*)sWT)[idx] = ((const uint4*)hBcT)[idx];
    for (int idx = tid; idx < 16 * DN; idx += 256) {
        int m = idx >> 7, k = idx & 127;
        sA[idx] = (_Float16)node_vec[(size_t)(m0 + m) * DN + k];
    }
    __syncthreads();

    const int nt = wv;
    const int f  = nt * 16 + (lane & 15);
    const int hi = lane >> 4;
    float bv = bcb[f];
    v8f acc;
#pragma unroll
    for (int r = 0; r < 8; ++r) acc[r] = bv;
#pragma unroll
    for (int kc = 0; kc < 4; ++kc)
        acc = WMMA_F16(frag_a_h(sA, DN, lane, kc * 32),
                       frag_b_h(sWT, DN, lane, nt * 16, kc * 32), acc);
#pragma unroll
    for (int r = 0; r < 8; ++r)
        X[(size_t)(m0 + r + 8 * hi) * DF + f] = acc[r];
}

// ---------------- kernel 2: filter net + cutoff/mask + aggregation + edge passthrough ----
__global__ __launch_bounds__(256) void k_edge(const float* __restrict__ edge_vec,
                                              const unsigned char* __restrict__ edge_mask,
                                              const float* __restrict__ edge_cutoff,
                                              const _Float16* __restrict__ hWT, // W1^T|W2^T contiguous
                                              const float* __restrict__ fb1,
                                              const float* __restrict__ fb2,
                                              const float* __restrict__ X,
                                              float* __restrict__ Yagg,
                                              float* __restrict__ out_edge) {
    __shared__ alignas(16) _Float16 sW[DF * DE + DF * DF]; // fW1^T (16KB) + fW2^T (32KB)
    __shared__ alignas(16) _Float16 sH[8 * 16 * DF];       // per-wave silu(H), 32KB
    __shared__ float sY[DF];
    _Float16* sW1 = sW;
    _Float16* sW2 = sW + DF * DE;

    const int blk = blockIdx.x;            // b*N + i
    const int b = blk >> 8;
    const int tid = threadIdx.x, lane = tid & 31, wv = tid >> 5;
    const int n = lane & 15, hi = lane >> 4;

    const float* Ebase = edge_vec + (size_t)blk * Nn * DE;

#if HAVE_TDM
    if (wv == 0)
        tdm_load_1d(lds_offset(sW), hWT, (unsigned)((DF * DE + DF * DF) * 2 / 8));
#else
    for (int idx = tid; idx < (DF * DE + DF * DF) / 8; idx += 256)
        ((uint4*)sW)[idx] = ((const uint4*)hWT)[idx];
#endif

    // Edge passthrough fused here (overlaps with TDM weight staging).
    {
        const float4* s4 = (const float4*)Ebase;
        float4* d4 = (float4*)(out_edge + (size_t)blk * Nn * DE);
        for (int idx = tid; idx < (Nn * DE) / 4; idx += 256) d4[idx] = s4[idx];
    }
    if (tid < DF) sY[tid] = 0.0f;
#if HAVE_TDM
    if (wv == 0) __builtin_amdgcn_s_wait_tensorcnt(0);
#endif
    __syncthreads();

    const float* Xb   = X + (size_t)b * Nn * DF;
    const float* cutb = edge_cutoff + (size_t)blk * Nn;
    const unsigned char* maskb = edge_mask + (size_t)blk * Nn;
    _Float16* myH = sH + wv * (16 * DF);

    for (int jb = wv; jb < 16; jb += 8) {
        const float* Ej = Ebase + (size_t)jb * 16 * DE;
        // GEMM1: H = silu(E @ fW1 + b1), K=64
        v16h a0 = frag_a_f32(Ej, DE, lane, 0);
        v16h a1 = frag_a_f32(Ej, DE, lane, 32);
#pragma unroll
        for (int nt = 0; nt < 8; ++nt) {
            const int f = nt * 16 + n;
            float bv = fb1[f];
            v8f acc;
#pragma unroll
            for (int r = 0; r < 8; ++r) acc[r] = bv;
            acc = WMMA_F16(a0, frag_b_h(sW1, DE, lane, nt * 16, 0), acc);
            acc = WMMA_F16(a1, frag_b_h(sW1, DE, lane, nt * 16, 32), acc);
#pragma unroll
            for (int r = 0; r < 8; ++r)
                myH[(r + 8 * hi) * DF + f] = (_Float16)silu_f(acc[r]);
        }
        float cm[8];
#pragma unroll
        for (int r = 0; r < 8; ++r) {
            int j = jb * 16 + r + 8 * hi;
            cm[r] = cutb[j] * (maskb[j] ? 1.0f : 0.0f);
        }
        // GEMM2: G = H @ fW2 + b2 (K=128), then aggregate over j
        v16h ha[4];
#pragma unroll
        for (int kc = 0; kc < 4; ++kc) ha[kc] = frag_a_h(myH, DF, lane, kc * 32);
#pragma unroll
        for (int nt = 0; nt < 8; ++nt) {
            const int f = nt * 16 + n;
            float bv = fb2[f];
            v8f acc;
#pragma unroll
            for (int r = 0; r < 8; ++r) acc[r] = bv;
#pragma unroll
            for (int kc = 0; kc < 4; ++kc)
                acc = WMMA_F16(ha[kc], frag_b_h(sW2, DF, lane, nt * 16, kc * 32), acc);
            float part = 0.0f;
#pragma unroll
            for (int r = 0; r < 8; ++r) {
                int j = jb * 16 + r + 8 * hi;
                part += (acc[r] * cm[r]) * Xb[(size_t)j * DF + f];
            }
            atomicAdd(&sY[f], part);   // ds_add_f32
        }
    }
    __syncthreads();
    if (tid < DF) Yagg[(size_t)blk * DF + tid] = sY[tid];
}

// ---------------- kernel 3: out = node_vec + silu(y@acW1+b1)@acW2+b2 ----------------
__global__ __launch_bounds__(256) void k_ac(const float* __restrict__ Yagg,
                                            const float* __restrict__ node_vec,
                                            const _Float16* __restrict__ hAc1T,
                                            const float* __restrict__ acb1,
                                            const _Float16* __restrict__ hAc2T,
                                            const float* __restrict__ acb2,
                                            float* __restrict__ out) {
    __shared__ alignas(16) _Float16 sW1T[DN * DF];
    __shared__ alignas(16) _Float16 sW2T[DN * DN];
    __shared__ alignas(16) _Float16 sA[16 * DF];
    __shared__ alignas(16) _Float16 sH2[16 * DN];
    const int tid = threadIdx.x, lane = tid & 31, wv = tid >> 5;
    const int m0 = blockIdx.x * 16;
    const int hi = lane >> 4;

    for (int idx = tid; idx < (DN * DF) / 8; idx += 256)
        ((uint4*)sW1T)[idx] = ((const uint4*)hAc1T)[idx];
    for (int idx = tid; idx < (DN * DN) / 8; idx += 256)
        ((uint4*)sW2T)[idx] = ((const uint4*)hAc2T)[idx];
    for (int idx = tid; idx < 16 * DF; idx += 256) {
        int m = idx >> 7, k = idx & 127;
        sA[idx] = (_Float16)Yagg[(size_t)(m0 + m) * DF + k];
    }
    __syncthreads();

    const int nt = wv;
    const int f  = nt * 16 + (lane & 15);
    {
        float bv = acb1[f];
        v8f acc;
#pragma unroll
        for (int r = 0; r < 8; ++r) acc[r] = bv;
#pragma unroll
        for (int kc = 0; kc < 4; ++kc)
            acc = WMMA_F16(frag_a_h(sA, DF, lane, kc * 32),
                           frag_b_h(sW1T, DF, lane, nt * 16, kc * 32), acc);
#pragma unroll
        for (int r = 0; r < 8; ++r)
            sH2[(r + 8 * hi) * DN + f] = (_Float16)silu_f(acc[r]);
    }
    __syncthreads();
    {
        float bv = acb2[f];
        v8f acc;
#pragma unroll
        for (int r = 0; r < 8; ++r) acc[r] = bv;
#pragma unroll
        for (int kc = 0; kc < 4; ++kc)
            acc = WMMA_F16(frag_a_h(sH2, DN, lane, kc * 32),
                           frag_b_h(sW2T, DN, lane, nt * 16, kc * 32), acc);
#pragma unroll
        for (int r = 0; r < 8; ++r) {
            size_t row = (size_t)(m0 + r + 8 * hi);
            out[row * DN + f] = acc[r] + node_vec[row * DN + f];
        }
    }
}

extern "C" void kernel_launch(void* const* d_in, const int* in_sizes, int n_in,
                              void* d_out, int out_size, void* d_ws, size_t ws_size,
                              hipStream_t stream) {
    (void)in_sizes; (void)n_in; (void)out_size; (void)ws_size;
    const float* node_vec    = (const float*)d_in[0];
    /* d_in[1] node_mask: unused by reference math */
    const float* edge_vec    = (const float*)d_in[2];
    const unsigned char* edge_mask = (const unsigned char*)d_in[3];
    const float* edge_cutoff = (const float*)d_in[4];
    const float* fW1 = (const float*)d_in[5];
    const float* fb1 = (const float*)d_in[6];
    const float* fW2 = (const float*)d_in[7];
    const float* fb2 = (const float*)d_in[8];
    const float* bcW = (const float*)d_in[9];
    const float* bcb = (const float*)d_in[10];
    const float* acW1 = (const float*)d_in[11];
    const float* acb1 = (const float*)d_in[12];
    const float* acW2 = (const float*)d_in[13];
    const float* acb2 = (const float*)d_in[14];

    // Workspace layout
    float* X    = (float*)d_ws;                      // [1024,128] fp32
    float* Yagg = X + ROWS * DF;                     // [1024,128] fp32
    _Float16* hW1T  = (_Float16*)(Yagg + ROWS * DF); // fW1^T  f16 [128][64]
    _Float16* hW2T  = hW1T + DF * DE;                // fW2^T  f16 [128][128] (contiguous after hW1T)
    _Float16* hBcT  = hW2T + DF * DF;                // bcW^T  f16 [128][128]
    _Float16* hAc1T = hBcT + DN * DF;                // acW1^T f16 [128][128]
    _Float16* hAc2T = hAc1T + DF * DN;               // acW2^T f16 [128][128]

    float* out_node = (float*)d_out;                 // [4,256,128]
    float* out_edge = out_node + ROWS * DN;          // [4,256,256,64]

    // Weight transpose/convert (re-run every call; deterministic).
    k_cvt_t<<<(DE * DF + 255) / 256, 256, 0, stream>>>(fW1, hW1T, DE, DF);
    k_cvt_t<<<(DF * DF + 255) / 256, 256, 0, stream>>>(fW2, hW2T, DF, DF);
    k_cvt_t<<<(DN * DF + 255) / 256, 256, 0, stream>>>(bcW, hBcT, DN, DF);
    k_cvt_t<<<(DF * DN + 255) / 256, 256, 0, stream>>>(acW1, hAc1T, DF, DN);
    k_cvt_t<<<(DN * DN + 255) / 256, 256, 0, stream>>>(acW2, hAc2T, DN, DN);

    k_bc<<<ROWS / 16, 256, 0, stream>>>(node_vec, hBcT, bcb, X);
    k_edge<<<Bn * Nn, 256, 0, stream>>>(edge_vec, edge_mask, edge_cutoff,
                                        hW1T, fb1, fb2, X, Yagg, out_edge);
    k_ac<<<ROWS / 16, 256, 0, stream>>>(Yagg, node_vec, hAc1T, acb1, hAc2T, acb2, out_node);
}
